// Recurrent_55585466745510
// MI455X (gfx1250) — matrix operand order repaired and checked by reading, not compiled, not run on hardware
//
#include <hip/hip_runtime.h>

typedef __attribute__((ext_vector_type(16))) __bf16 v16bf;
typedef __attribute__((ext_vector_type(8)))  float  v8f;

#define BB 64
#define TT 256
#define DD 1024
#define HH 1024
#define G4 4096   /* 4*H */

// ---------------------------------------------------------------- helpers
__device__ __forceinline__ __bf16 f2bf(float f) {
    unsigned u = __builtin_bit_cast(unsigned, f);
    unsigned r = (u + 0x7FFFu + ((u >> 16) & 1u)) >> 16;   // RNE
    unsigned short s = (unsigned short)r;
    return __builtin_bit_cast(__bf16, s);
}

__device__ __forceinline__ float sigmoidf_(float x) {
    return 1.0f / (1.0f + expf(-x));
}

__device__ __forceinline__ v8f splat8(float v) {
    v8f r;
    #pragma unroll
    for (int i = 0; i < 8; ++i) r[i] = v;
    return r;
}

// ---------------------------------------------------------------- GEMM
// gates[64,4096] = bias + A0[64,1024] @ W[:,0:1024]^T + A1[64,1024] @ W[:,1024:2048]^T
// W packed row-major [4096][2048] bf16.
// Each wave owns a 64x32 output block: 4 M-tiles x 2 N-tiles.
//   A fragment reused x2 (both N tiles), B fragment reused x4 (all M tiles).
// Grid: 16 WGs x 8 waves -> 128 waves cover N=4096; W streamed once per step.
__global__ __launch_bounds__(256)
void lstm_gemm(const __bf16* __restrict__ A0, long strideA0,
               const __bf16* __restrict__ A1,
               const __bf16* __restrict__ W,
               const float* __restrict__ bias,
               float* __restrict__ gates)
{
    const int tid  = threadIdx.x;
    const int wave = tid >> 5;
    const int lane = tid & 31;
    const int l15  = lane & 15;
    const int kh   = lane >> 4;                 // K-half select
    const int n0   = blockIdx.x * 256 + wave * 32;

    const __bf16* __restrict__ Wr0 = W + (size_t)(n0 + l15) * 2048;
    const __bf16* __restrict__ Wr1 = W + (size_t)(n0 + 16 + l15) * 2048;

    const __bf16* Ar0[4];
    const __bf16* Ar1[4];
    #pragma unroll
    for (int mt = 0; mt < 4; ++mt) {
        const int m = mt * 16 + l15;
        Ar0[mt] = A0 + (size_t)m * strideA0;
        Ar1[mt] = A1 + (size_t)m * 1024;
    }

    // C/D layout: all 8 accumulator rows share one column -> bias folds into init.
    v8f acc[4][2];
    {
        const float b0 = bias[n0 + l15];
        const float b1 = bias[n0 + 16 + l15];
        #pragma unroll
        for (int mt = 0; mt < 4; ++mt) {
            acc[mt][0] = splat8(b0);
            acc[mt][1] = splat8(b1);
        }
    }

    // ---- K half 1: A0 vs W cols [0,1024)
    #pragma unroll 2
    for (int k0 = 0; k0 < 1024; k0 += 32) {
        v16bf bf0 = *(const v16bf*)(Wr0 + k0 + 16 * kh);
        v16bf bf1 = *(const v16bf*)(Wr1 + k0 + 16 * kh);
        #pragma unroll
        for (int mt = 0; mt < 4; ++mt) {
            v16bf a;
            ((uint4*)&a)[0] = *(const uint4*)(Ar0[mt] + k0 + 16 * kh);
            ((uint4*)&a)[1] = *(const uint4*)(Ar0[mt] + k0 + 16 * kh + 8);
            acc[mt][0] = __builtin_amdgcn_wmma_f32_16x16x32_bf16(
                             false, a, false, bf0, (short)0, acc[mt][0], false, false);
            acc[mt][1] = __builtin_amdgcn_wmma_f32_16x16x32_bf16(
                             false, a, false, bf1, (short)0, acc[mt][1], false, false);
        }
    }
    // ---- K half 2: A1 vs W cols [1024,2048)
    #pragma unroll 2
    for (int k0 = 0; k0 < 1024; k0 += 32) {
        v16bf bf0 = *(const v16bf*)(Wr0 + 1024 + k0 + 16 * kh);
        v16bf bf1 = *(const v16bf*)(Wr1 + 1024 + k0 + 16 * kh);
        #pragma unroll
        for (int mt = 0; mt < 4; ++mt) {
            v16bf a;
            ((uint4*)&a)[0] = *(const uint4*)(Ar1[mt] + k0 + 16 * kh);
            ((uint4*)&a)[1] = *(const uint4*)(Ar1[mt] + k0 + 16 * kh + 8);
            acc[mt][0] = __builtin_amdgcn_wmma_f32_16x16x32_bf16(
                             false, a, false, bf0, (short)0, acc[mt][0], false, false);
            acc[mt][1] = __builtin_amdgcn_wmma_f32_16x16x32_bf16(
                             false, a, false, bf1, (short)0, acc[mt][1], false, false);
        }
    }

    // store: C/D v8f -> row = mt*16 + 8*kh + r, col = n0 + 16*j + l15
    #pragma unroll
    for (int mt = 0; mt < 4; ++mt) {
        const int row0 = mt * 16 + 8 * kh;
        #pragma unroll
        for (int j = 0; j < 2; ++j) {
            const int col = n0 + 16 * j + l15;
            #pragma unroll
            for (int r = 0; r < 8; ++r)
                gates[(size_t)(row0 + r) * G4 + col] = acc[mt][j][r];
        }
    }
}

// ---------------------------------------------------------------- LSTM cell
// gates already include bias (folded into WMMA accumulator init).
__global__ __launch_bounds__(256)
void lstm_cell(const float* __restrict__ gates,
               float* __restrict__ c, __bf16* __restrict__ hb,
               float* __restrict__ out_h, long out_stride,
               float* __restrict__ hN, float* __restrict__ cN, int write_final)
{
    int idx = blockIdx.x * 256 + threadIdx.x;      // 65536 = 64*1024
    int b   = idx >> 10;
    int n   = idx & 1023;
    const float* g = gates + (size_t)b * G4;

    float i  = sigmoidf_(g[n]);
    float f  = sigmoidf_(g[1024 + n]);
    float gt = tanhf    (g[2048 + n]);
    float o  = sigmoidf_(g[3072 + n]);

    float cn = f * c[idx] + i * gt;
    float hn = o * tanhf(cn);

    c[idx]  = cn;
    hb[idx] = f2bf(hn);
    if (out_h)        out_h[(size_t)b * out_stride + n] = hn;
    if (write_final) { hN[idx] = hn; cN[idx] = cn; }
}

// ---------------------------------------------------------------- converts
__global__ void f32_to_bf16_k(const float* __restrict__ src,
                              __bf16* __restrict__ dst, long n)
{
    long i = (long)blockIdx.x * blockDim.x + threadIdx.x;
    long stride = (long)gridDim.x * blockDim.x;
    for (; i < n; i += stride) dst[i] = f2bf(src[i]);
}

// Pack [Wih[l] | Whh[l]] rows into bf16 W[l][4096][2048]
__global__ void pack_w_k(const float* __restrict__ Wih,
                         const float* __restrict__ Whh,
                         __bf16* __restrict__ Wpk)
{
    long total = 2L * G4 * 2048;
    long i = (long)blockIdx.x * blockDim.x + threadIdx.x;
    long stride = (long)gridDim.x * blockDim.x;
    for (; i < total; i += stride) {
        long l   = i >> 23;            // / (4096*2048)
        long rem = i & 8388607L;
        long row = rem >> 11;          // / 2048
        long k   = rem & 2047L;
        float v = (k < 1024)
                ? Wih[(l * G4 + row) * 1024 + k]
                : Whh[(l * G4 + row) * 1024 + (k - 1024)];
        Wpk[i] = f2bf(v);
    }
}

__global__ void init_state_k(const float* __restrict__ h0,
                             const float* __restrict__ c0,
                             __bf16* __restrict__ hb,
                             float* __restrict__ c)
{
    int i = blockIdx.x * blockDim.x + threadIdx.x;   // 131072 = 2*64*1024
    if (i < 2 * BB * HH) { hb[i] = f2bf(h0[i]); c[i] = c0[i]; }
}

// ---------------------------------------------------------------- launch
extern "C" void kernel_launch(void* const* d_in, const int* in_sizes, int n_in,
                              void* d_out, int out_size, void* d_ws, size_t ws_size,
                              hipStream_t stream) {
    const float* x    = (const float*)d_in[0];   // [64,256,1024]
    const float* h0   = (const float*)d_in[1];   // [2,64,1024]
    const float* c0   = (const float*)d_in[2];   // [2,64,1024]
    const float* Wih  = (const float*)d_in[3];   // [2,4096,1024]
    const float* Whh  = (const float*)d_in[4];   // [2,4096,1024]
    const float* bias = (const float*)d_in[5];   // [2,4096]
    float* out = (float*)d_out;

    // workspace carve-up (~69 MB)
    __bf16* xb    = (__bf16*)d_ws;                       // 16,777,216 bf16
    __bf16* Wpk   = xb  + (size_t)BB * TT * DD;          // 16,777,216 bf16
    __bf16* hb    = Wpk + 2L * G4 * 2048;                // 131,072 bf16
    float*  cst   = (float*)(hb + 2 * BB * HH);          // 131,072 f32
    float*  gates = cst + 2 * BB * HH;                   // 262,144 f32

    __bf16* W0  = Wpk;
    __bf16* W1  = Wpk + (size_t)G4 * 2048;
    __bf16* h0b = hb;
    __bf16* h1b = hb + BB * HH;
    float*  c0f = cst;
    float*  c1f = cst + BB * HH;

    float* hN = out + (size_t)BB * TT * HH;   // [2,64,1024]
    float* cN = hN + 2L * BB * HH;

    // one-time conversions (inside the call: deterministic, capture-safe)
    f32_to_bf16_k<<<4096, 256, 0, stream>>>(x, xb, (long)BB * TT * DD);
    pack_w_k     <<<4096, 256, 0, stream>>>(Wih, Whh, Wpk);
    init_state_k <<<512,  256, 0, stream>>>(h0, c0, hb, cst);

    for (int t = 0; t < TT; ++t) {
        int fin = (t == TT - 1) ? 1 : 0;
        // layer 0: A = [x_t | h0_prev]
        lstm_gemm<<<16, 256, 0, stream>>>(xb + (size_t)t * DD, (long)TT * DD,
                                          h0b, W0, bias, gates);
        lstm_cell<<<256, 256, 0, stream>>>(gates, c0f, h0b,
                                           nullptr, 0, hN, cN, fin);
        // layer 1: A = [h0_t | h1_prev]
        lstm_gemm<<<16, 256, 0, stream>>>(h0b, 1024L, h1b, W1, bias + G4, gates);
        lstm_cell<<<256, 256, 0, stream>>>(gates, c1f, h1b,
                                           out + (size_t)t * HH, (long)TT * HH,
                                           hN + BB * HH, cN + BB * HH, fin);
    }
}
